// PlacementNetV2_88433376625029
// MI455X (gfx1250) — matrix operand order repaired and checked by reading, not compiled, not run on hardware
//
#include <hip/hip_runtime.h>
#include <hip/hip_bf16.h>

// ---------------------------------------------------------------------------
// PlacementNetV2 for MI455X (gfx1250): bf16 WMMA GEMMs, fp32 atomic graph agg.
// ---------------------------------------------------------------------------

typedef __attribute__((ext_vector_type(16))) __bf16 v16bf;
typedef __attribute__((ext_vector_type(8)))  float  v8f;

__device__ __forceinline__ unsigned short f2bf(float f) {
  union { float f; unsigned u; } c; c.f = f;
  unsigned u = c.u;
  u += 0x7fffu + ((u >> 16) & 1u);   // round-to-nearest-even
  return (unsigned short)(u >> 16);
}
__device__ __forceinline__ float bf2f(unsigned short h) {
  union { unsigned u; float f; } c; c.u = ((unsigned)h) << 16;
  return c.f;
}

union FragU { uint4 q[2]; v16bf v; };

__device__ __forceinline__ v8f wmma_bf16(v16bf a, v16bf b, v8f c) {
  // D = A(16x32) * B(32x16) + C, fp32 accum
  return __builtin_amdgcn_wmma_f32_16x16x32_bf16(false, a, false, b, (short)0, c,
                                                 false, false);
}

// A-fragment (16x32 bf16) from row-major [rows, ld] bf16 storage.
// ISA layout: lanes 0-15 -> M=lane, K in {kb..kb+7, kb+16..kb+23} with kb=0;
//             lanes 16-31 -> same M, kb=8.  Two contiguous 16B loads per lane.
__device__ __forceinline__ v16bf load_a16(const unsigned short* A, int ld,
                                          int rowBase, int kstep, int lane) {
  int m  = lane & 15;
  int kb = kstep * 32 + ((lane & 16) ? 8 : 0);
  const unsigned short* p = A + (size_t)(rowBase + m) * ld + kb;
  FragU f;
  f.q[0] = *(const uint4*)(p);
  f.q[1] = *(const uint4*)(p + 16);
  return f.v;
}

// B-fragment from pre-packed buffer: block = 32 lanes x 8 dwords (1KB).
__device__ __forceinline__ v16bf load_b16(const unsigned* P, int blk, int lane) {
  const uint4* p = (const uint4*)(P + (size_t)blk * 256 + (size_t)lane * 8);
  FragU f; f.q[0] = p[0]; f.q[1] = p[1];
  return f.v;
}

// ---------------------------------------------------------------------------
// K0: pack a (batch of) fp32 weight matrices [Kdim, Ndim] into bf16 B-fragment
// layout.  One block per (batch, ntile, kblk); thread tid -> lane=tid/8, j=tid%8.
// B layout: lane<16 -> K {0..7,16..23} of the 32-K block, lane>=16 -> K {8..15,24..31};
// dword j holds K = base+2j', base+2j'+1 (j'=j&3, +16 if j>=4) for column n = ntile*16 + (lane&15).
// ---------------------------------------------------------------------------
__global__ void pack_weights(const float* __restrict__ W, unsigned* __restrict__ dst,
                             int Kdim, int Ndim, int nKblk, int perBatchBlocks,
                             long long srcStride, long long dstStride) {
  int bid = blockIdx.x;
  int b   = bid / perBatchBlocks;
  int rem = bid % perBatchBlocks;
  int ntile = rem / nKblk;
  int kblk  = rem % nKblk;
  int tid  = threadIdx.x;
  int lane = tid >> 3;
  int j    = tid & 7;
  int k0 = kblk * 32 + ((lane & 16) ? 8 : 0) + 2 * (j & 3) + ((j & 4) ? 16 : 0);
  int n  = ntile * 16 + (lane & 15);
  const float* Wb = W + (size_t)b * srcStride;
  float w0 = (k0 < Kdim     && n < Ndim) ? Wb[(size_t)k0 * Ndim + n]       : 0.f;
  float w1 = (k0 + 1 < Kdim && n < Ndim) ? Wb[(size_t)(k0 + 1) * Ndim + n] : 0.f;
  unsigned pk = (unsigned)f2bf(w0) | ((unsigned)f2bf(w1) << 16);
  dst[(size_t)b * dstStride + (size_t)rem * 256 + lane * 8 + j] = pk;
}

// ---------------------------------------------------------------------------
// K1: layer-1 aggregation: agg1[dst,0:17] += x[src,0:17]; deg[dst] += 1
// ---------------------------------------------------------------------------
__global__ void sage1_agg(const float* __restrict__ x, const long long* __restrict__ ei,
                          float* __restrict__ agg1, float* __restrict__ deg, int E) {
  long long stride = (long long)gridDim.x * blockDim.x;
  for (long long e = (long long)blockIdx.x * blockDim.x + threadIdx.x; e < E; e += stride) {
    __builtin_prefetch(&ei[e + 2048], 0, 0);               // global_prefetch_b8
    int s = (int)ei[e];
    int d = (int)ei[(size_t)E + e];
    const float* xs = x + (size_t)s * 17;
    float* ad = agg1 + (size_t)d * 32;
    atomicAdd(&deg[d], 1.0f);
#pragma unroll
    for (int f = 0; f < 17; ++f) atomicAdd(&ad[f], xs[f]);
  }
}

// K2: aggn1 = bf16(agg1/deg) padded to 32 cols; xpad = bf16(x) padded.
__global__ void sage1_norm(const float* __restrict__ agg1, const float* __restrict__ deg,
                           const float* __restrict__ x,
                           unsigned short* __restrict__ aggn1,
                           unsigned short* __restrict__ xpad, int N) {
  int n = blockIdx.x * blockDim.x + threadIdx.x;
  if (n >= N) return;
  float invd = 1.0f / fmaxf(deg[n], 1.0f);
#pragma unroll
  for (int c = 0; c < 32; ++c) {
    float a  = (c < 17) ? agg1[(size_t)n * 32 + c] * invd : 0.f;
    float xv = (c < 17) ? x[(size_t)n * 17 + c] : 0.f;
    aggn1[(size_t)n * 32 + c] = f2bf(a);
    xpad [(size_t)n * 32 + c] = f2bf(xv);
  }
}

// K3: h1 = relu(bn1(aggn1@W1l + xpad@W1r + b1)), bf16 out. 8 waves, 1 K-step.
__global__ void sage1_gemm(const unsigned short* __restrict__ aggn1,
                           const unsigned short* __restrict__ xpad,
                           const unsigned* __restrict__ Wl, const unsigned* __restrict__ Wr,
                           const float* __restrict__ bias, const float* __restrict__ g,
                           const float* __restrict__ bb,  const float* __restrict__ mu,
                           const float* __restrict__ var, unsigned short* __restrict__ h1) {
  int rowBase = blockIdx.x * 16;
  int wave = threadIdx.x >> 5, lane = threadIdx.x & 31;
  v8f c = {};
  c = wmma_bf16(load_a16(aggn1, 32, rowBase, 0, lane), load_b16(Wl, wave, lane), c);
  c = wmma_bf16(load_a16(xpad,  32, rowBase, 0, lane), load_b16(Wr, wave, lane), c);
  int col = wave * 16 + (lane & 15);
  float bsv = bias[col], mm = mu[col], bbv = bb[col];
  float rs = rsqrtf(var[col] + 1e-5f) * g[col];
#pragma unroll
  for (int r = 0; r < 8; ++r) {
    int rr = (lane & 16) ? r + 8 : r;
    float val = c[r] + bsv;
    val = (val - mm) * rs + bbv;           // bn then relu (matches reference)
    val = fmaxf(val, 0.f);
    h1[(size_t)(rowBase + rr) * 128 + col] = f2bf(val);
  }
}

// K5: layer-2 aggregation: agg2[dst,:] += h1[src,:] ; 4 threads per edge.
__global__ void sage2_agg(const unsigned short* __restrict__ h1,
                          const long long* __restrict__ ei,
                          float* __restrict__ agg2, int E) {
  long long total  = (long long)E * 4;
  long long stride = (long long)gridDim.x * blockDim.x;
  for (long long t = (long long)blockIdx.x * blockDim.x + threadIdx.x; t < total; t += stride) {
    long long e = t >> 2;
    int chunk = (int)(t & 3);
    __builtin_prefetch(&ei[e + 1024], 0, 0);
    int s = (int)ei[e];
    int d = (int)ei[(size_t)E + e];
    const uint4* hq = (const uint4*)(h1 + (size_t)s * 128 + chunk * 32);
    float* ad = agg2 + (size_t)d * 128 + chunk * 32;
    uint4 q0 = hq[0], q1 = hq[1], q2 = hq[2], q3 = hq[3];
    unsigned w[16] = {q0.x,q0.y,q0.z,q0.w, q1.x,q1.y,q1.z,q1.w,
                      q2.x,q2.y,q2.z,q2.w, q3.x,q3.y,q3.z,q3.w};
#pragma unroll
    for (int i = 0; i < 16; ++i) {
      atomicAdd(&ad[2 * i + 0], bf2f((unsigned short)(w[i] & 0xffffu)));
      atomicAdd(&ad[2 * i + 1], bf2f((unsigned short)(w[i] >> 16)));
    }
  }
}

// K6: aggn2 = bf16(agg2/deg). 4 threads per node.
__global__ void sage2_norm(const float* __restrict__ agg2, const float* __restrict__ deg,
                           unsigned short* __restrict__ aggn2, int N) {
  int t = blockIdx.x * blockDim.x + threadIdx.x;
  int n = t >> 2, chunk = t & 3;
  if (n >= N) return;
  float invd = 1.0f / fmaxf(deg[n], 1.0f);
  const float* a = agg2 + (size_t)n * 128 + chunk * 32;
  unsigned short* o = aggn2 + (size_t)n * 128 + chunk * 32;
#pragma unroll
  for (int f = 0; f < 32; ++f) o[f] = f2bf(a[f] * invd);
}

// K7: h2 = relu(bn2(aggn2@W2l + h1@W2r + b2)), bf16 out. 8 waves x 4 K-steps.
__global__ void sage2_gemm(const unsigned short* __restrict__ aggn2,
                           const unsigned short* __restrict__ h1,
                           const unsigned* __restrict__ Wl, const unsigned* __restrict__ Wr,
                           const float* __restrict__ bias, const float* __restrict__ g,
                           const float* __restrict__ bb,  const float* __restrict__ mu,
                           const float* __restrict__ var, unsigned short* __restrict__ h2) {
  int rowBase = blockIdx.x * 16;
  int wave = threadIdx.x >> 5, lane = threadIdx.x & 31;
  v8f c = {};
#pragma unroll
  for (int kk = 0; kk < 4; ++kk) {
    c = wmma_bf16(load_a16(aggn2, 128, rowBase, kk, lane), load_b16(Wl, wave * 4 + kk, lane), c);
    c = wmma_bf16(load_a16(h1,    128, rowBase, kk, lane), load_b16(Wr, wave * 4 + kk, lane), c);
  }
  int col = wave * 16 + (lane & 15);
  float bsv = bias[col], mm = mu[col], bbv = bb[col];
  float rs = rsqrtf(var[col] + 1e-5f) * g[col];
#pragma unroll
  for (int r = 0; r < 8; ++r) {
    int rr = (lane & 16) ? r + 8 : r;
    float val = c[r] + bsv;
    val = (val - mm) * rs + bbv;
    val = fmaxf(val, 0.f);
    h2[(size_t)(rowBase + rr) * 128 + col] = f2bf(val);
  }
}

// K8: count head. 4 waves: z = bn(relu(h2@W1+b1)) [16x64] via WMMA -> LDS,
// then counts = z@W2 + b2 (tiny, VALU) -> out cols 0..5.
__global__ void count_head(const unsigned short* __restrict__ h2,
                           const unsigned* __restrict__ Wp,
                           const float* __restrict__ b1,  const float* __restrict__ g,
                           const float* __restrict__ bb,  const float* __restrict__ mu,
                           const float* __restrict__ var,
                           const float* __restrict__ W2,  const float* __restrict__ b2,
                           float* __restrict__ out) {
  __shared__ float zs[16][64];
  int rowBase = blockIdx.x * 16;
  int wave = threadIdx.x >> 5, lane = threadIdx.x & 31;
  v8f c = {};
#pragma unroll
  for (int kk = 0; kk < 4; ++kk)
    c = wmma_bf16(load_a16(h2, 128, rowBase, kk, lane), load_b16(Wp, wave * 4 + kk, lane), c);
  int col = wave * 16 + (lane & 15);
  float bsv = b1[col], mm = mu[col], bbv = bb[col];
  float rs = rsqrtf(var[col] + 1e-5f) * g[col];
#pragma unroll
  for (int r = 0; r < 8; ++r) {
    int rr = (lane & 16) ? r + 8 : r;
    float val = fmaxf(c[r] + bsv, 0.f);    // relu then bn (matches reference)
    zs[rr][col] = (val - mm) * rs + bbv;
  }
  __syncthreads();
  int t = threadIdx.x;
  if (t < 96) {
    int mrow = t / 6, j = t % 6;
    float acc = b2[j];
#pragma unroll 8
    for (int kk = 0; kk < 64; ++kk) acc += zs[mrow][kk] * W2[kk * 6 + j];
    out[(size_t)(rowBase + mrow) * 150 + j] = acc;
  }
}

// K9: 6 fused coord heads per 16-row tile. 8 waves.
// stage1: [16x128] WMMA -> relu -> bn -> bf16 LDS
// stage2: [16x64]  WMMA (A from LDS) -> relu -> bn -> bf16 LDS
// stage3: [16x24]  WMMA -> sigmoid -> out cols 6 + k*24 + q
__global__ void coord_heads(const unsigned short* __restrict__ h2,
                            const unsigned* __restrict__ W1p, const unsigned* __restrict__ W2p,
                            const unsigned* __restrict__ W3p,
                            const float* __restrict__ b1, const float* __restrict__ g1,
                            const float* __restrict__ bb1, const float* __restrict__ m1,
                            const float* __restrict__ v1,
                            const float* __restrict__ b2, const float* __restrict__ g2,
                            const float* __restrict__ bb2, const float* __restrict__ m2,
                            const float* __restrict__ v2,
                            const float* __restrict__ b3, float* __restrict__ out) {
  __shared__ unsigned short z1[16 * 128];
  __shared__ unsigned short z2[16 * 64];
  int rowBase = blockIdx.x * 16;
  int wave = threadIdx.x >> 5, lane = threadIdx.x & 31;
  for (int k = 0; k < 6; ++k) {
    {   // stage 1: all 8 waves, 128 output cols
      v8f c = {};
#pragma unroll
      for (int kk = 0; kk < 4; ++kk)
        c = wmma_bf16(load_a16(h2, 128, rowBase, kk, lane),
                      load_b16(W1p, k * 32 + wave * 4 + kk, lane), c);
      int col = wave * 16 + (lane & 15);
      int o = k * 128 + col;
      float bsv = b1[o], mm = m1[o], bbv = bb1[o];
      float rs = rsqrtf(v1[o] + 1e-5f) * g1[o];
#pragma unroll
      for (int r = 0; r < 8; ++r) {
        int rr = (lane & 16) ? r + 8 : r;
        float val = fmaxf(c[r] + bsv, 0.f);
        z1[rr * 128 + col] = f2bf((val - mm) * rs + bbv);
      }
    }
    __syncthreads();
    if (wave < 4) {   // stage 2: 64 output cols, A from LDS
      v8f c = {};
#pragma unroll
      for (int kk = 0; kk < 4; ++kk)
        c = wmma_bf16(load_a16(z1, 128, 0, kk, lane),
                      load_b16(W2p, k * 16 + wave * 4 + kk, lane), c);
      int col = wave * 16 + (lane & 15);
      int o = k * 64 + col;
      float bsv = b2[o], mm = m2[o], bbv = bb2[o];
      float rs = rsqrtf(v2[o] + 1e-5f) * g2[o];
#pragma unroll
      for (int r = 0; r < 8; ++r) {
        int rr = (lane & 16) ? r + 8 : r;
        float val = fmaxf(c[r] + bsv, 0.f);
        z2[rr * 64 + col] = f2bf((val - mm) * rs + bbv);
      }
    }
    __syncthreads();
    if (wave < 2) {   // stage 3: 24 (padded 32) output cols
      v8f c = {};
#pragma unroll
      for (int kk = 0; kk < 2; ++kk)
        c = wmma_bf16(load_a16(z2, 64, 0, kk, lane),
                      load_b16(W3p, k * 4 + wave * 2 + kk, lane), c);
      int col = wave * 16 + (lane & 15);
      if (col < 24) {
        float bsv = b3[k * 24 + col];
#pragma unroll
        for (int r = 0; r < 8; ++r) {
          int rr = (lane & 16) ? r + 8 : r;
          float val = c[r] + bsv;
          val = 1.0f / (1.0f + __expf(-val));
          out[(size_t)(rowBase + rr) * 150 + 6 + k * 24 + col] = val;
        }
      }
    }
    __syncthreads();
  }
}

// ---------------------------------------------------------------------------
extern "C" void kernel_launch(void* const* d_in, const int* in_sizes, int n_in,
                              void* d_out, int out_size, void* d_ws, size_t ws_size,
                              hipStream_t stream) {
  const float*     x  = (const float*)d_in[0];
  const long long* ei = (const long long*)d_in[1];   // int64 edge_index
  const int N = in_sizes[0] / 17;
  const int E = in_sizes[1] / 2;

  const float* W1l = (const float*)d_in[2];
  const float* W1r = (const float*)d_in[3];
  const float* b1  = (const float*)d_in[4];
  const float* bn1g = (const float*)d_in[5], *bn1b = (const float*)d_in[6];
  const float* bn1m = (const float*)d_in[7], *bn1v = (const float*)d_in[8];
  const float* W2l = (const float*)d_in[9];
  const float* W2r = (const float*)d_in[10];
  const float* b2  = (const float*)d_in[11];
  const float* bn2g = (const float*)d_in[12], *bn2b = (const float*)d_in[13];
  const float* bn2m = (const float*)d_in[14], *bn2v = (const float*)d_in[15];
  const float* cW1 = (const float*)d_in[16], *cb1 = (const float*)d_in[17];
  const float* cbg = (const float*)d_in[18], *cbb = (const float*)d_in[19];
  const float* cbm = (const float*)d_in[20], *cbv = (const float*)d_in[21];
  const float* cW2 = (const float*)d_in[22], *cb2 = (const float*)d_in[23];
  const float* chW1 = (const float*)d_in[24], *chb1 = (const float*)d_in[25];
  const float* chg1 = (const float*)d_in[26], *chbb1 = (const float*)d_in[27];
  const float* chm1 = (const float*)d_in[28], *chv1 = (const float*)d_in[29];
  const float* chW2 = (const float*)d_in[30], *chb2 = (const float*)d_in[31];
  const float* chg2 = (const float*)d_in[32], *chbb2 = (const float*)d_in[33];
  const float* chm2 = (const float*)d_in[34], *chv2 = (const float*)d_in[35];
  const float* chW3 = (const float*)d_in[36], *chb3 = (const float*)d_in[37];

  float* out = (float*)d_out;

  // ---- workspace carve-up (256B aligned) ----
  char* ws = (char*)d_ws;
  size_t off = 0;
  auto carve = [&](size_t bytes) -> char* {
    char* p = ws + off;
    off = (off + bytes + 255) & ~(size_t)255;
    return p;
  };
  float*          deg   = (float*)carve((size_t)N * 4);
  float*          agg1  = (float*)carve((size_t)N * 32 * 4);
  unsigned short* aggn1 = (unsigned short*)carve((size_t)N * 32 * 2);
  unsigned short* xpad  = (unsigned short*)carve((size_t)N * 32 * 2);
  unsigned short* h1    = (unsigned short*)carve((size_t)N * 128 * 2);
  float*          agg2  = (float*)carve((size_t)N * 128 * 4);
  unsigned short* aggn2 = (unsigned short*)carve((size_t)N * 128 * 2);
  unsigned short* h2    = (unsigned short*)carve((size_t)N * 128 * 2);
  unsigned* pW1l  = (unsigned*)carve(8   * 1024);
  unsigned* pW1r  = (unsigned*)carve(8   * 1024);
  unsigned* pW2l  = (unsigned*)carve(32  * 1024);
  unsigned* pW2r  = (unsigned*)carve(32  * 1024);
  unsigned* pCnt  = (unsigned*)carve(16  * 1024);
  unsigned* pCh1  = (unsigned*)carve(192 * 1024);
  unsigned* pCh2  = (unsigned*)carve(96  * 1024);
  unsigned* pCh3  = (unsigned*)carve(24  * 1024);
  (void)ws_size; (void)n_in; (void)out_size;

  // ---- weight packing (bf16 B-fragment layout) ----
  pack_weights<<<8,   256, 0, stream>>>(W1l, pW1l, 17, 128, 1, 8,  0, 0);
  pack_weights<<<8,   256, 0, stream>>>(W1r, pW1r, 17, 128, 1, 8,  0, 0);
  pack_weights<<<32,  256, 0, stream>>>(W2l, pW2l, 128, 128, 4, 32, 0, 0);
  pack_weights<<<32,  256, 0, stream>>>(W2r, pW2r, 128, 128, 4, 32, 0, 0);
  pack_weights<<<16,  256, 0, stream>>>(cW1, pCnt, 128, 64, 4, 16, 0, 0);
  pack_weights<<<192, 256, 0, stream>>>(chW1, pCh1, 128, 128, 4, 32, 128 * 128, 32 * 256);
  pack_weights<<<96,  256, 0, stream>>>(chW2, pCh2, 128, 64, 4, 16, 128 * 64, 16 * 256);
  pack_weights<<<24,  256, 0, stream>>>(chW3, pCh3, 64, 24, 2, 4, 64 * 24, 4 * 256);

  // ---- zero accumulators ----
  hipMemsetAsync(deg,  0, (size_t)N * 4,        stream);
  hipMemsetAsync(agg1, 0, (size_t)N * 32 * 4,   stream);
  hipMemsetAsync(agg2, 0, (size_t)N * 128 * 4,  stream);

  const int rowTiles = N / 16;   // N = 100000 -> 6250, exact

  // ---- SAGE layer 1 ----
  sage1_agg<<<4096, 256, 0, stream>>>(x, ei, agg1, deg, E);
  sage1_norm<<<(N + 255) / 256, 256, 0, stream>>>(agg1, deg, x, aggn1, xpad, N);
  sage1_gemm<<<rowTiles, 256, 0, stream>>>(aggn1, xpad, pW1l, pW1r,
                                           b1, bn1g, bn1b, bn1m, bn1v, h1);

  // ---- SAGE layer 2 ----
  sage2_agg<<<8192, 256, 0, stream>>>(h1, ei, agg2, E);
  sage2_norm<<<(N * 4 + 255) / 256, 256, 0, stream>>>(agg2, deg, aggn2, N);
  sage2_gemm<<<rowTiles, 256, 0, stream>>>(aggn2, h1, pW2l, pW2r,
                                           b2, bn2g, bn2b, bn2m, bn2v, h2);

  // ---- heads ----
  count_head<<<rowTiles, 128, 0, stream>>>(h2, pCnt, cb1, cbg, cbb, cbm, cbv,
                                           cW2, cb2, out);
  coord_heads<<<rowTiles, 256, 0, stream>>>(h2, pCh1, pCh2, pCh3,
                                            chb1, chg1, chbb1, chm1, chv1,
                                            chb2, chg2, chbb2, chm2, chv2,
                                            chb3, out);
}
